// ParallelModel2_37804302139686
// MI455X (gfx1250) — matrix operand order repaired and checked by reading, not compiled, not run on hardware
//
#include <hip/hip_runtime.h>

typedef __attribute__((ext_vector_type(16))) _Float16 v16h;
typedef __attribute__((ext_vector_type(8)))  float    v8f;

#define BB   16384
#define TT   128
#define FF   29
#define KK   512
#define NTW  16          // N-tiles per wave (2 waves cover N=512)
#define TPB  2           // M-tiles per block
#define WPB  (TPB * 2)   // 2 waves per tile -> 4 waves / 128 threads per block

__device__ __forceinline__ float fast_tanh(float x) {
#if __has_builtin(__builtin_amdgcn_tanhf)
    return __builtin_amdgcn_tanhf(x);
#elif __has_builtin(__builtin_amdgcn_tanh_f32)
    return __builtin_amdgcn_tanh_f32(x);
#else
    float ax = __builtin_fabsf(x);
    float e  = __builtin_amdgcn_exp2f(-2.8853900817779268f * ax);
    float t  = (1.0f - e) * __builtin_amdgcn_rcpf(1.0f + e);
    return __builtin_copysignf(t, x);
#endif
}

// A-matrix K index held in VGPR j (halves 2j, 2j+1) for this lane half.
__device__ __forceinline__ int a_k0(int j, int hi) {
    return (j < 4 ? 2 * j : 8 + 2 * j) + 8 * hi;
}

__global__ __launch_bounds__(32 * WPB, 1)
void narx_wmma_kernel(const float* __restrict__ X,
                      const float* __restrict__ Y0,
                      const float* __restrict__ W1,
                      const float* __restrict__ B1,
                      const float* __restrict__ W2,
                      const float* __restrict__ B2,
                      float* __restrict__ OUT) {
    __shared__ float spred[TPB][2][16];

    const int lane = threadIdx.x & 31;
    const int w    = threadIdx.x >> 5;
    const int pair = w >> 1;            // which M-tile in this block
    const int half = w & 1;             // which N half this wave owns
    const int tile = blockIdx.x * TPB + pair;
    const int r    = lane & 15;
    const int hi   = lane >> 4;
    const int b    = tile * 16 + r;
    const int nb   = half * NTW;        // first N-tile for this wave

    // ---- Preload this wave's half of W1 in WMMA B-layout (K = a_k0(j,hi), +1) ----
    v16h  w1r[NTW];
    float b1v[NTW];
    float w2v[NTW];
#pragma unroll
    for (int n = 0; n < NTW; ++n) {
        const int col = (nb + n) * 16 + r;
        v16h bm;
#pragma unroll
        for (int j = 0; j < 8; ++j) {
            const int k0 = a_k0(j, hi);
            bm[2 * j]     = (_Float16)W1[(size_t)k0 * KK + col];
            bm[2 * j + 1] = (_Float16)W1[(size_t)(k0 + 1) * KK + col];
        }
        w1r[n] = bm;
        b1v[n] = B1[col];
        w2v[n] = W2[col];
    }
    const float b2s = B2[0];

    float fb0 = Y0[(size_t)b * 3 + 0];
    float fb1 = Y0[(size_t)b * 3 + 1];
    float fb2 = Y0[(size_t)b * 3 + 2];

    // ---- Double-buffered per-lane X gather (clamped so k>=29 slots stay in-bounds) ----
    float xc[16], xn[16];
    {
        const float* xr = X + (size_t)b * TT * FF;
#pragma unroll
        for (int j = 0; j < 8; ++j) {
            const int k0 = a_k0(j, hi);
            const int k1 = k0 + 1;
            xc[2 * j]     = xr[k0 > 28 ? 28 : k0];
            xc[2 * j + 1] = xr[k1 > 28 ? 28 : k1];
        }
    }

    const v8f czero = {};

    for (int t = 0; t < TT; ++t) {
        // ---- Build A tile: substitute feedback into K = 29..31 slots ----
        v16h a;
#pragma unroll
        for (int j = 0; j < 8; ++j) {
            const int k0 = a_k0(j, hi);
            const int k1 = k0 + 1;
            const float v0 = (k0 <= 28) ? xc[2 * j]     : ((k0 == 29) ? fb0 : ((k0 == 30) ? fb1 : fb2));
            const float v1 = (k1 <= 28) ? xc[2 * j + 1] : ((k1 == 29) ? fb0 : ((k1 == 30) ? fb1 : fb2));
            a[2 * j]     = (_Float16)v0;
            a[2 * j + 1] = (_Float16)v1;
        }

        // ---- Prefetch next step's X gather before the compute burst ----
        if (t + 1 < TT) {
            const float* xr = X + ((size_t)b * TT + (t + 1)) * FF;
#pragma unroll
            for (int j = 0; j < 8; ++j) {
                const int k0 = a_k0(j, hi);
                const int k1 = k0 + 1;
                xn[2 * j]     = xr[k0 > 28 ? 28 : k0];
                xn[2 * j + 1] = xr[k1 > 28 ? 28 : k1];
            }
        }

        // ---- Software-pipelined WMMA chain: next WMMA issues before prior drain ----
        float pacc[8];
#pragma unroll
        for (int v = 0; v < 8; ++v) pacc[v] = 0.0f;

        v8f cA = __builtin_amdgcn_wmma_f32_16x16x32_f16(
                     false, a, false, w1r[0], (short)0, czero, false, false);
#pragma unroll
        for (int nn = 0; nn < NTW - 1; ++nn) {
            v8f cB = __builtin_amdgcn_wmma_f32_16x16x32_f16(
                         false, a, false, w1r[nn + 1], (short)0, czero, false, false);
            const float b1s = b1v[nn], w2s = w2v[nn];
#pragma unroll
            for (int v = 0; v < 8; ++v)
                pacc[v] = __builtin_fmaf(fast_tanh(cA[v] + b1s), w2s, pacc[v]);
            cA = cB;
        }
        {
            const float b1s = b1v[NTW - 1], w2s = w2v[NTW - 1];
#pragma unroll
            for (int v = 0; v < 8; ++v)
                pacc[v] = __builtin_fmaf(fast_tanh(cA[v] + b1s), w2s, pacc[v]);
        }

        // ---- Reduce over 16 lanes (columns) within each half ----
        float pr[8];
#pragma unroll
        for (int v = 0; v < 8; ++v) {
            float s = pacc[v];
            s += __shfl_xor(s, 1, 32);
            s += __shfl_xor(s, 2, 32);
            s += __shfl_xor(s, 4, 32);
            s += __shfl_xor(s, 8, 32);
            pr[v] = s;
        }

        // ---- Combine the two waves' N-halves through LDS ----
        if (r == 0) {
#pragma unroll
            for (int v = 0; v < 8; ++v) spred[pair][half][hi * 8 + v] = pr[v];
        }
        __syncthreads();
        const float pred_r = spred[pair][0][r] + spred[pair][1][r] + b2s;
        if ((lane < 16) && (half == 0)) OUT[(size_t)b * TT + t] = pred_r;
        fb2 = fb1; fb1 = fb0; fb0 = pred_r;
        __syncthreads();

#pragma unroll
        for (int i = 0; i < 16; ++i) xc[i] = xn[i];
    }
}

extern "C" void kernel_launch(void* const* d_in, const int* in_sizes, int n_in,
                              void* d_out, int out_size, void* d_ws, size_t ws_size,
                              hipStream_t stream) {
    const float* X  = (const float*)d_in[0];
    const float* Y0 = (const float*)d_in[1];
    const float* W1 = (const float*)d_in[2];
    const float* B1 = (const float*)d_in[3];
    const float* W2 = (const float*)d_in[4];
    const float* B2 = (const float*)d_in[5];
    float* OUT = (float*)d_out;

    const int tiles  = BB / 16;          // 1024 M-tiles
    const int blocks = tiles / TPB;      // 2 tiles/block, 2 waves/tile
    narx_wmma_kernel<<<blocks, 32 * WPB, 0, stream>>>(X, Y0, W1, B1, W2, B2, OUT);
}
